// AttentionHead_31688268710459
// MI455X (gfx1250) — compile-verified, matrix operand order checked by reading, not run on hardware
//
#include <hip/hip_runtime.h>

#define B_  8
#define S_  4096
#define D_  1024
#define DK_ 64

typedef __attribute__((ext_vector_type(16))) _Float16 v16h;
typedef __attribute__((ext_vector_type(8)))  _Float16 h8;
typedef __attribute__((ext_vector_type(4)))  _Float16 h4;
typedef __attribute__((ext_vector_type(8)))  float    v8f;
typedef __attribute__((ext_vector_type(4)))  int      v4i;

// pointer types matching the async builtin's parameters
typedef __attribute__((address_space(1))) v4i* as1_v4i_ptr;   // global
typedef __attribute__((address_space(3))) v4i* as3_v4i_ptr;   // LDS

// ---- feature probes (device pass only) -------------------------------------
#if defined(__AMDGCN__) && __has_builtin(__builtin_amdgcn_global_load_async_to_lds_b128)
#  define HAS_ASYNC_LDS 1
#  pragma message("PROBE: __builtin_amdgcn_global_load_async_to_lds_b128 AVAILABLE")
#else
#  define HAS_ASYNC_LDS 0
#  if defined(__AMDGCN__)
#    pragma message("PROBE: __builtin_amdgcn_global_load_async_to_lds_b128 MISSING")
#  endif
#endif
#if defined(__AMDGCN__) && __has_builtin(__builtin_amdgcn_s_wait_asynccnt)
#  define HAS_WAIT_ASYNC 1
#else
#  define HAS_WAIT_ASYNC 0
#endif

// 16-bit A/B matrix VGPR layout (CDNA5 ISA §7.12.2)
__device__ __forceinline__ int pair_base(int lane, int p) {
  int kb = (lane & 16) ? 8 : 0;
  return (p < 4) ? (kb + 2 * p) : (16 + kb + 2 * (p - 4));
}

__device__ __forceinline__ v8f wmma_f16(v16h a, v16h b, v8f c) {
  return __builtin_amdgcn_wmma_f32_16x16x32_f16(
      false, a, false, b, (short)0, c, false, false);
}

// Stage one 16B chunk of K (row-major block) and one of Vt (dk-major block).
__device__ __forceinline__ void stage_pair(const _Float16* kg, const _Float16* vg,
                                           _Float16* kl, _Float16* vl) {
#if HAS_ASYNC_LDS
  __builtin_amdgcn_global_load_async_to_lds_b128(
      (as1_v4i_ptr)kg, (as3_v4i_ptr)kl, 0, 0);
  __builtin_amdgcn_global_load_async_to_lds_b128(
      (as1_v4i_ptr)vg, (as3_v4i_ptr)vl, 0, 0);
#else
  *(uint4*)kl = *(const uint4*)kg;
  *(uint4*)vl = *(const uint4*)vg;
#endif
}

// ---------------------------------------------------------------------------
// Kernel 1: Q/K = x @ W^T (row-major f16 out), V stored TRANSPOSED Vt[b][dk][s].
// ---------------------------------------------------------------------------
__global__ __launch_bounds__(256) void qkv_proj_kernel(
    const float* __restrict__ x,
    const float* __restrict__ wq,
    const float* __restrict__ wk,
    const float* __restrict__ wv,
    _Float16* __restrict__ Qo,
    _Float16* __restrict__ Ko,
    _Float16* __restrict__ Vt)
{
  const int tid  = threadIdx.x;
  const int wave = tid >> 5;
  const int lane = tid & 31;
  const int lo   = lane & 15;
  const int hi   = lane >> 4;
  const int row0 = blockIdx.x * 128;

  __shared__ __align__(16) _Float16 xt[128][32];
  __shared__ __align__(16) _Float16 wt[3][64][32];

  v8f acc[3][4];
#pragma unroll
  for (int pj = 0; pj < 3; ++pj)
#pragma unroll
    for (int nt = 0; nt < 4; ++nt)
      acc[pj][nt] = (v8f){0.f,0.f,0.f,0.f,0.f,0.f,0.f,0.f};

  const int am = wave * 16 + lo;
  const int sr = tid >> 3;          // staging row 0..31
  const int sc = (tid & 7) * 4;     // staging col 0,4,...,28

  for (int kc = 0; kc < D_ / 32; ++kc) {
    const int k0 = kc * 32;
    // stage x tile [128][32] fp32 -> f16 via float4 loads / h4 LDS stores
#pragma unroll
    for (int j = 0; j < 4; ++j) {
      const int r = sr + j * 32;
      const float4 f = *(const float4*)&x[(size_t)(row0 + r) * D_ + (k0 + sc)];
      h4 hv = {(_Float16)f.x, (_Float16)f.y, (_Float16)f.z, (_Float16)f.w};
      *(h4*)&xt[r][sc] = hv;
    }
    // stage weight tiles [64][32] x3
#pragma unroll
    for (int j = 0; j < 2; ++j) {
      const int r = sr + j * 32;
      const float4 fq = *(const float4*)&wq[(size_t)r * D_ + (k0 + sc)];
      const float4 fk = *(const float4*)&wk[(size_t)r * D_ + (k0 + sc)];
      const float4 fv = *(const float4*)&wv[(size_t)r * D_ + (k0 + sc)];
      h4 hq = {(_Float16)fq.x, (_Float16)fq.y, (_Float16)fq.z, (_Float16)fq.w};
      h4 hk = {(_Float16)fk.x, (_Float16)fk.y, (_Float16)fk.z, (_Float16)fk.w};
      h4 hv = {(_Float16)fv.x, (_Float16)fv.y, (_Float16)fv.z, (_Float16)fv.w};
      *(h4*)&wt[0][r][sc] = hq;
      *(h4*)&wt[1][r][sc] = hk;
      *(h4*)&wt[2][r][sc] = hv;
    }
    if (kc + 1 < D_ / 32)   // prefetch next x chunk (global_prefetch_b8)
      __builtin_prefetch(&x[(size_t)(row0 + sr) * D_ + (k0 + 32 + sc)], 0, 0);
    __syncthreads();

    v16h a;
#pragma unroll
    for (int p = 0; p < 8; ++p) {
      int k = pair_base(lane, p);
      a[2*p]   = xt[am][k];
      a[2*p+1] = xt[am][k+1];
    }
#pragma unroll
    for (int pj = 0; pj < 3; ++pj) {
#pragma unroll
      for (int nt = 0; nt < 4; ++nt) {
        v16h bm;
#pragma unroll
        for (int p = 0; p < 8; ++p) {
          int k = pair_base(lane, p);
          bm[2*p]   = wt[pj][nt*16 + lo][k];
          bm[2*p+1] = wt[pj][nt*16 + lo][k+1];
        }
        acc[pj][nt] = wmma_f16(a, bm, acc[pj][nt]);
      }
    }
    __syncthreads();
  }

  // Q, K row-major stores
  _Float16* outs[2] = {Qo, Ko};
#pragma unroll
  for (int pj = 0; pj < 2; ++pj)
#pragma unroll
    for (int nt = 0; nt < 4; ++nt)
#pragma unroll
      for (int r = 0; r < 8; ++r) {
        int row = row0 + wave * 16 + r + 8 * hi;
        outs[pj][(size_t)row * DK_ + nt * 16 + lo] = (_Float16)acc[pj][nt][r];
      }

  // V transposed store: rows r are contiguous along S -> one b128 per n-tile
  const int bb  = row0 / S_;
  const int sr0 = (row0 - bb * S_) + wave * 16 + 8 * hi;  // first of 8 seq rows
#pragma unroll
  for (int nt = 0; nt < 4; ++nt) {
    h8 pk;
#pragma unroll
    for (int r = 0; r < 8; ++r) pk[r] = (_Float16)acc[2][nt][r];
    *(h8*)&Vt[((size_t)bb * DK_ + nt * 16 + lo) * S_ + sr0] = pk;
  }
}

// ---------------------------------------------------------------------------
// Kernel 2: causal flash attention, double-buffered async K/V staging.
// ---------------------------------------------------------------------------
__global__ __launch_bounds__(256) void flash_attn_kernel(
    const _Float16* __restrict__ Q,
    const _Float16* __restrict__ K,
    const _Float16* __restrict__ Vt,
    float* __restrict__ O)
{
  const int tid   = threadIdx.x;
  const int wave  = __builtin_amdgcn_readfirstlane(tid >> 5);
  const int lane  = tid & 31;
  const int lo    = lane & 15;
  const int hi    = lane >> 4;
  const int bidx  = blockIdx.y;
  const int qblk  = blockIdx.x;

  const _Float16* Qb  = Q  + (size_t)bidx * S_ * DK_;
  const _Float16* Kb  = K  + (size_t)bidx * S_ * DK_;
  const _Float16* Vtb = Vt + (size_t)bidx * DK_ * S_;
  float* Ob = O + (size_t)bidx * S_ * DK_;

  const int q0 = qblk * 128 + wave * 16;   // wave-uniform
  const int qh = q0 + 15;

  __shared__ __align__(16) _Float16 Kl [2][32][64];
  __shared__ __align__(16) _Float16 Vtl[2][64][32];
  __shared__ __align__(16) _Float16 Pl [8][16][32];

  // Q A-tiles
  v16h qa[2];
#pragma unroll
  for (int kc = 0; kc < 2; ++kc)
#pragma unroll
    for (int p = 0; p < 8; ++p) {
      int k = kc * 32 + pair_base(lane, p);
      const _Float16* qr = Qb + (size_t)(q0 + lo) * DK_;
      qa[kc][2*p]   = qr[k];
      qa[kc][2*p+1] = qr[k+1];
    }

  v8f oacc[4];
#pragma unroll
  for (int nt = 0; nt < 4; ++nt)
    oacc[nt] = (v8f){0.f,0.f,0.f,0.f,0.f,0.f,0.f,0.f};
  float ms[8], ls[8];
#pragma unroll
  for (int r = 0; r < 8; ++r) { ms[r] = -3.0e38f; ls[r] = 0.f; }

  // per-thread staging coordinates (one 16B chunk per matrix per block)
  const int koff = tid * 8;            // halfs into K block [32][64]
  const int vrow = tid >> 2;           // dk row 0..63
  const int vcol = (tid & 3) * 8;      // col chunk in [0,32)

  const float SCL = 0.03125f * 1.44269504088896340736f;  // d_model^-0.5 * log2(e)
  const int nkb = qblk * 4 + 4;

  stage_pair(Kb + koff, Vtb + (size_t)vrow * S_ + vcol,
             &Kl[0][0][0] + koff, &Vtl[0][vrow][vcol]);

  for (int kb = 0; kb < nkb; ++kb) {
    const int cur = kb & 1;
    const int t0  = kb * 32;
    const bool have_next = (kb + 1 < nkb);   // block-uniform
    if (have_next) {
      const int t1 = t0 + 32;
      stage_pair(Kb + (size_t)t1 * DK_ + koff,
                 Vtb + (size_t)vrow * S_ + t1 + vcol,
                 &Kl[cur ^ 1][0][0] + koff, &Vtl[cur ^ 1][vrow][vcol]);
    }
#if HAS_ASYNC_LDS
#  if HAS_WAIT_ASYNC
    if (have_next) __builtin_amdgcn_s_wait_asynccnt(2);
    else           __builtin_amdgcn_s_wait_asynccnt(0);
#  else
    if (have_next) asm volatile("s_wait_asynccnt 0x2" ::: "memory");
    else           asm volatile("s_wait_asynccnt 0x0" ::: "memory");
#  endif
#endif
    __syncthreads();

    const bool active = (t0 <= qh);   // wave-uniform -> EXEC all-1s inside
    v16h pa;
    if (active) {
      v8f s0 = (v8f){0.f,0.f,0.f,0.f,0.f,0.f,0.f,0.f};
      v8f s1 = (v8f){0.f,0.f,0.f,0.f,0.f,0.f,0.f,0.f};
#pragma unroll
      for (int kc = 0; kc < 2; ++kc) {
        v16h b0, b1;   // B[k][n] = K[t0+coloff+n][k]
#pragma unroll
        for (int p = 0; p < 8; ++p) {
          int k = kc * 32 + pair_base(lane, p);
          b0[2*p]   = Kl[cur][lo][k];       b0[2*p+1] = Kl[cur][lo][k+1];
          b1[2*p]   = Kl[cur][16 + lo][k];  b1[2*p+1] = Kl[cur][16 + lo][k+1];
        }
        s0 = wmma_f16(qa[kc], b0, s0);
        s1 = wmma_f16(qa[kc], b1, s1);
      }

      float alpha[8];
#pragma unroll
      for (int r = 0; r < 8; ++r) {
        const int qrow = q0 + r + 8 * hi;
        float x0 = s0[r] * SCL;             // base-2 domain
        float x1 = s1[r] * SCL;
        if (t0 + lo > qrow)      x0 = -3.0e38f;
        if (t0 + 16 + lo > qrow) x1 = -3.0e38f;
        float rm = fmaxf(x0, x1);
        rm = fmaxf(rm, __shfl_xor(rm, 1, 32));
        rm = fmaxf(rm, __shfl_xor(rm, 2, 32));
        rm = fmaxf(rm, __shfl_xor(rm, 4, 32));
        rm = fmaxf(rm, __shfl_xor(rm, 8, 32));
        float mn = fmaxf(ms[r], rm);
        float al = __builtin_amdgcn_exp2f(ms[r] - mn);
        float p0 = __builtin_amdgcn_exp2f(x0 - mn);
        float p1 = __builtin_amdgcn_exp2f(x1 - mn);
        float rs = p0 + p1;
        rs += __shfl_xor(rs, 1, 32);
        rs += __shfl_xor(rs, 2, 32);
        rs += __shfl_xor(rs, 4, 32);
        rs += __shfl_xor(rs, 8, 32);
        ls[r] = ls[r] * al + rs;
        ms[r] = mn;
        alpha[r] = al;
        Pl[wave][r + 8*hi][lo]      = (_Float16)p0;
        Pl[wave][r + 8*hi][16 + lo] = (_Float16)p1;
      }
#pragma unroll
      for (int r = 0; r < 8; ++r) {
        oacc[0][r] *= alpha[r]; oacc[1][r] *= alpha[r];
        oacc[2][r] *= alpha[r]; oacc[3][r] *= alpha[r];
      }
    }
    __syncthreads();

    if (active) {
#pragma unroll
      for (int p = 0; p < 8; ++p) {
        int k = pair_base(lane, p);
        pa[2*p]   = Pl[wave][lo][k];
        pa[2*p+1] = Pl[wave][lo][k+1];
      }
#pragma unroll
      for (int nt = 0; nt < 4; ++nt) {
        v16h bv;   // B[k][n] = Vt[nt*16+n][t0+k] -> contiguous pairs along k
#pragma unroll
        for (int p = 0; p < 8; ++p) {
          int k = pair_base(lane, p);
          bv[2*p]   = Vtl[cur][nt*16 + lo][k];
          bv[2*p+1] = Vtl[cur][nt*16 + lo][k+1];
        }
        oacc[nt] = wmma_f16(pa, bv, oacc[nt]);
      }
    }
    __syncthreads();
  }

  // epilogue: O = acc / l (fp32)
#pragma unroll
  for (int r = 0; r < 8; ++r) {
    const int row = q0 + r + 8 * hi;
    const float inv = 1.0f / ls[r];
#pragma unroll
    for (int nt = 0; nt < 4; ++nt)
      Ob[(size_t)row * DK_ + nt * 16 + lo] = oacc[nt][r] * inv;
  }
}

// ---------------------------------------------------------------------------
extern "C" void kernel_launch(void* const* d_in, const int* in_sizes, int n_in,
                              void* d_out, int out_size, void* d_ws, size_t ws_size,
                              hipStream_t stream) {
  const float* x  = (const float*)d_in[0];
  const float* wq = (const float*)d_in[1];
  const float* wk = (const float*)d_in[2];
  const float* wv = (const float*)d_in[3];
  float* out = (float*)d_out;

  _Float16* qws = (_Float16*)d_ws;
  _Float16* kws = qws + (size_t)B_ * S_ * DK_;
  _Float16* vws = kws + (size_t)B_ * S_ * DK_;   // transposed V: [B][DK][S]

  qkv_proj_kernel<<<dim3((B_ * S_) / 128), 256, 0, stream>>>(
      x, wq, wk, wv, qws, kws, vws);
  flash_attn_kernel<<<dim3(S_ / 128, B_), 256, 0, stream>>>(
      qws, kws, vws, out);
}
